// GridNerfNet_45672682226378
// MI455X (gfx1250) — matrix operand order repaired and checked by reading, not compiled, not run on hardware
//
#include <hip/hip_runtime.h>
#include <cstdint>

typedef __attribute__((ext_vector_type(16))) _Float16 v16h;
typedef __attribute__((ext_vector_type(2)))  _Float16 h2;
typedef __attribute__((ext_vector_type(8)))  float    v8f;
typedef __attribute__((ext_vector_type(2)))  float    f2;

// ---- hash-grid constants (computed from the reference config) ----
__device__ __constant__ int c_res[16] = {16,23,31,43,59,81,112,154,213,295,407,562,777,1073,1483,2048};
__device__ __constant__ int c_off[16] = {0,4920,18744,51512,136696,352696,876984,1401272,
                                         1925560,2449848,2974136,3498424,4022712,4547000,
                                         5071288,5595576};
// levels >= 5 are hashed, all with table size 2^19 (mask 0x7FFFF)

// packed fp16 weight workspace layout (halfs):
//   L0: 3 Ktiles x 16 Ntiles x 512  = 24576   @ 0
//   L1: 8 x 16 x 512                = 65536   @ 24576
//   L2: 8 x 17 x 512                = 69632   @ 90112
//   total 159744 halfs  (= 319488 bytes)

// softplus(100x)*0.01 computed entirely in the log2 domain so the scale
// constants fold into the transcendentals (v_exp_f32/v_log_f32 are base-2):
//   0.01*ln(1+e^{100x}) = 0.01*ln2 * ( max(t,0) + log2(1 + 2^{-|t|}) ),
//   t = 100*log2(e) * x
__device__ __forceinline__ float softplus100(float x) {
  const float t = 144.26950408889634f * x;                       // 100*log2(e)
  const float e = __builtin_amdgcn_exp2f(-fabsf(t));             // v_exp_f32
  const float r = fmaxf(t, 0.f) + __builtin_amdgcn_logf(1.f + e);// v_log_f32
  return 0.0069314718055994531f * r;                             // 0.01*ln(2)
}

// Load A fragment (16x32 f16) from a wave-private LDS slice [16 rows x 256 cols].
// ISA layout: lanes 0-15 -> M=lane, lanes 16-31 -> M=lane-16 with K offset +8;
// VGPR v holds K pair { (v<4?0:16) + hi*8 + (v&3)*2 , +1 }.
// The 8-half runs are contiguous -> compiler emits 2x ds_load_b128 per fragment.
__device__ __forceinline__ v16h load_a(const _Float16* h, int kt, int lane) {
  const int row = lane & 15;
  const int hi  = (lane >> 4) << 3;  // 0 or 8
  v16h a;
#pragma unroll
  for (int v = 0; v < 8; ++v) {
    const int k = kt * 32 + ((v & 4) ? 16 : 0) + hi + ((v & 3) << 1);
    h2 p = *(const h2*)(h + row * 256 + k);
    a[2 * v]     = p[0];
    a[2 * v + 1] = p[1];
  }
  return a;
}

template <int KT, int NT, bool FINAL>
__device__ __forceinline__ void mlp_layer(const _Float16* __restrict__ wfrag,
                                          const _Float16* hin,
                                          const float* __restrict__ bias, int dout,
                                          _Float16* hout, float* gout, int lane) {
  const int ncol = lane & 15;
  const int mhi  = (lane >> 4) << 3;  // C/D layout: VGPR r -> M = r + mhi

  // A fragments are invariant across the N-tile loop: register-block them once
  // (KT=8 -> 64 VGPRs) instead of re-reading LDS per tile.
  v16h afrag[KT];
#pragma unroll
  for (int kt = 0; kt < KT; ++kt) afrag[kt] = load_a(hin, kt, lane);

#pragma unroll 2
  for (int nt = 0; nt < NT; ++nt) {
    v8f c = {0.f, 0.f, 0.f, 0.f, 0.f, 0.f, 0.f, 0.f};
#pragma unroll
    for (int kt = 0; kt < KT; ++kt) {
      v16h b = *(const v16h*)(wfrag + (size_t)(kt * NT + nt) * 512 + lane * 16);
      c = __builtin_amdgcn_wmma_f32_16x16x32_f16(false, afrag[kt], false, b,
                                                 (short)0, c, false, false);
    }
    const int   col = nt * 16 + ncol;
    const float bv  = (col < dout) ? bias[col] : 0.f;
#pragma unroll
    for (int r = 0; r < 8; ++r) {
      const int   mrow = mhi + r;
      const float val  = c[r] + bv;
      if (FINAL) {
        if (col < 257) gout[(size_t)mrow * 257 + col] = val;
      } else {
        hout[mrow * 256 + col] = (_Float16)softplus100(val);
      }
    }
  }
}

__global__ __launch_bounds__(128) void fused_mlp(
    const float* __restrict__ x, const float* __restrict__ table,
    const _Float16* __restrict__ wp,
    const float* __restrict__ b0, const float* __restrict__ b1,
    const float* __restrict__ b2, float* __restrict__ out) {
  // two ping-pong buffers, 4 waves x [16 x 256] fp16 each = 64 KB total
  __shared__ _Float16 sb[2 * 4 * 16 * 256];
  const int tid  = threadIdx.x;
  const int wave = tid >> 5;
  const int lane = tid & 31;

  // Warm L2 for the packed weights (global_prefetch_b8); 319488 B / 128B lines
  // = 2496 lines, ~20 per thread. Speculative; cheap relative to encoding.
  {
    const char* wb = (const char*)wp;
#pragma unroll 4
    for (int i = tid; i < 2496; i += 128) __builtin_prefetch(wb + i * 128, 0, 1);
  }

  // ---------- encoding: 2 threads per point (8 hash levels each) ----------
  {
    const int pl   = tid & 63;
    const int half = tid >> 6;
    const int g    = blockIdx.x * 64 + pl;
    const float x0 = x[3 * g + 0], x1 = x[3 * g + 1], x2 = x[3 * g + 2];
    _Float16* hb = &sb[(pl >> 4) * 4096 + (pl & 15) * 256];

    if (half == 0) {
      hb[0] = (_Float16)x0; hb[1] = (_Float16)x1; hb[2] = (_Float16)x2;
#pragma unroll
      for (int f = 0; f < 6; ++f) {
        const float s  = (float)(1 << f);
        const float a0 = x0 * s, a1 = x1 * s, a2 = x2 * s;
        hb[3 + f * 6 + 0] = (_Float16)__sinf(a0);
        hb[3 + f * 6 + 1] = (_Float16)__sinf(a1);
        hb[3 + f * 6 + 2] = (_Float16)__sinf(a2);
        hb[3 + f * 6 + 3] = (_Float16)__cosf(a0);
        hb[3 + f * 6 + 4] = (_Float16)__cosf(a1);
        hb[3 + f * 6 + 5] = (_Float16)__cosf(a2);
      }
    } else {
      for (int cpad = 71; cpad < 96; ++cpad) hb[cpad] = (_Float16)0.f;
    }

    const int lbase = half * 8;
    for (int l = lbase; l < lbase + 8; ++l) {
      const int   res = c_res[l];
      const float rm  = (float)(res - 1);
      const float px = x0 * rm, py = x1 * rm, pz = x2 * rm;
      const float fx = floorf(px), fy = floorf(py), fz = floorf(pz);
      const float wx = px - fx, wy = py - fy, wz = pz - fz;
      const uint32_t ix = (uint32_t)fx, iy = (uint32_t)fy, iz = (uint32_t)fz;
      const uint32_t st = (uint32_t)(res + 1);
      float f0 = 0.f, f1 = 0.f;
#pragma unroll
      for (int cidx = 0; cidx < 8; ++cidx) {
        const uint32_t ci = (cidx >> 2) & 1, cj = (cidx >> 1) & 1, ck = cidx & 1;
        const uint32_t cx = ix + ci, cy = iy + cj, cz = iz + ck;
        uint32_t idx;
        if (l >= 5)
          idx = (cx ^ (cy * 2654435761u) ^ (cz * 805459861u)) & 0x7FFFFu;
        else
          idx = cx + cy * st + cz * st * st;
        const f2 ft = *(const f2*)(table + 2u * ((uint32_t)c_off[l] + idx));
        const float wgt = (ci ? wx : 1.f - wx) * (cj ? wy : 1.f - wy) * (ck ? wz : 1.f - wz);
        f0 = fmaf(wgt, ft[0], f0);
        f1 = fmaf(wgt, ft[1], f1);
      }
      hb[39 + 2 * l]     = (_Float16)f0;
      hb[39 + 2 * l + 1] = (_Float16)f1;
    }
  }
  __syncthreads();

  // ---------- fused MLP: each wave owns its 16-point M-tile ----------
  _Float16* bufA = &sb[wave * 4096];          // holds h0 (cols 0..95 valid)
  _Float16* bufB = &sb[16384 + wave * 4096];

  mlp_layer<3, 16, false>(wp,          bufA, b0, 256, bufB, nullptr, lane);
  __syncthreads();
  mlp_layer<8, 16, false>(wp + 24576,  bufB, b1, 256, bufA, nullptr, lane);
  __syncthreads();
  float* gout = out + (size_t)(blockIdx.x * 64 + wave * 16) * 257;
  mlp_layer<8, 17, true>(wp + 90112,   bufA, b2, 257, nullptr, gout, lane);
}

// ---------- weight-norm scale precompute ----------
__global__ void prep_scale(const float* v0, const float* g0, const float* v1,
                           const float* g1, const float* v2, const float* g2,
                           float* scales) {
  const int i = blockIdx.x * blockDim.x + threadIdx.x;
  if (i >= 769) return;
  const float *v, *g; int din, row; float* dst;
  if (i < 256)      { v = v0; g = g0; din = 71;  row = i;       dst = scales;       }
  else if (i < 512) { v = v1; g = g1; din = 256; row = i - 256; dst = scales + 256; }
  else              { v = v2; g = g2; din = 256; row = i - 512; dst = scales + 512; }
  float acc = 0.f;
  for (int k = 0; k < din; ++k) { const float t = v[row * din + k]; acc = fmaf(t, t, acc); }
  dst[row] = g[row] / sqrtf(acc);
}

// ---------- pack normalized weights into fp16 B-fragment layout ----------
__global__ void pack_weights(const float* v0, const float* v1, const float* v2,
                             const float* scales, _Float16* wp) {
  const int i = blockIdx.x * blockDim.x + threadIdx.x;
  if (i >= 159744) return;
  int rel, NT, din, dout; const float* v; const float* sc;
  if (i < 24576)      { rel = i;          NT = 16; din = 71;  dout = 256; v = v0; sc = scales;       }
  else if (i < 90112) { rel = i - 24576;  NT = 16; din = 256; dout = 256; v = v1; sc = scales + 256; }
  else                { rel = i - 90112;  NT = 17; din = 256; dout = 257; v = v2; sc = scales + 512; }
  const int frag = rel >> 9;
  const int e    = rel & 511;
  const int lane = e >> 4;
  const int ee   = e & 15;
  const int kt = frag / NT, nt = frag % NT;
  const int N = nt * 16 + (lane & 15);
  const int K = kt * 32 + ((lane >> 4) << 3) + (ee & 7) + ((ee >> 3) << 4);
  const float val = (K < din && N < dout) ? v[N * din + K] * sc[N] : 0.f;
  wp[i] = (_Float16)val;
}

extern "C" void kernel_launch(void* const* d_in, const int* in_sizes, int n_in,
                              void* d_out, int out_size, void* d_ws, size_t ws_size,
                              hipStream_t stream) {
  (void)in_sizes; (void)n_in; (void)out_size; (void)ws_size;
  const float* x     = (const float*)d_in[0];
  const float* table = (const float*)d_in[1];
  const float* v0 = (const float*)d_in[2];
  const float* g0 = (const float*)d_in[3];
  const float* b0 = (const float*)d_in[4];
  const float* v1 = (const float*)d_in[5];
  const float* g1 = (const float*)d_in[6];
  const float* b1 = (const float*)d_in[7];
  const float* v2 = (const float*)d_in[8];
  const float* g2 = (const float*)d_in[9];
  const float* b2 = (const float*)d_in[10];

  float*    scales = (float*)d_ws;
  _Float16* wp     = (_Float16*)((char*)d_ws + 4096);
  float*    out    = (float*)d_out;

  prep_scale  <<<13,   64, 0, stream>>>(v0, g0, v1, g1, v2, g2, scales);
  pack_weights<<<624, 256, 0, stream>>>(v0, v1, v2, scales, wp);
  fused_mlp   <<<8192, 128, 0, stream>>>(x, table, wp, b0, b1, b2, out);
}
